// My_LSTM_40913858462326
// MI455X (gfx1250) — matrix operand order repaired and checked by reading, not compiled, not run on hardware
//
#include <hip/hip_runtime.h>

#define B_   1024
#define T_   256
#define D_   512
#define H_   512
#define KTOT 1024   // D + H
#define G4H  2048   // 4*H

typedef __attribute__((ext_vector_type(16))) __bf16       v16bf;
typedef __attribute__((ext_vector_type(8)))  float        v8f;
typedef __attribute__((ext_vector_type(4)))  unsigned int u32x4;
typedef __attribute__((ext_vector_type(4)))  float        f32x4;
typedef __attribute__((ext_vector_type(4)))  __bf16       bf16x4;

union Frag { u32x4 q[2]; v16bf v; };

__device__ __forceinline__ float sigmoidf_(float x) { return 1.0f / (1.0f + __expf(-x)); }

// LDS byte offset of a generic pointer into shared memory (addrspace(3) is 32-bit)
__device__ __forceinline__ unsigned lds_offset(const void* p) {
    return (unsigned)(unsigned long long)(__attribute__((address_space(3))) const void*)p;
}

// CDNA5 async global->LDS copy, 16 bytes per lane (tracked by ASYNCcnt)
__device__ __forceinline__ void async_load_b128(unsigned lds_off, const void* gaddr) {
    asm volatile("global_load_async_to_lds_b128 %0, %1, off"
                 :: "v"(lds_off), "v"(gaddr) : "memory");
}
__device__ __forceinline__ void wait_asynccnt0() {
    asm volatile("s_wait_asynccnt 0x0" ::: "memory");
}

// ---------------------------------------------------------------------------
// Pack Wg [4H, D+H] fp32 row-major  ->  bf16 [K/32][4H][32] (B-fragment native)
// ---------------------------------------------------------------------------
__global__ __launch_bounds__(256) void pack_wg_kernel(const float* __restrict__ Wg,
                                                      __bf16* __restrict__ out) {
    size_t e = (size_t)blockIdx.x * 256 + threadIdx.x;   // 2048*1024 elements
    int ki = (int)(e & 31);
    int kt = (int)((e >> 5) & 31);
    int n  = (int)(e >> 10);
    out[((size_t)kt * G4H + n) * 32 + ki] = (__bf16)Wg[(size_t)n * KTOT + kt * 32 + ki];
}

// Pack W1 [H/2, H] fp32 -> bf16 [K/32][256][32]
__global__ __launch_bounds__(256) void pack_w1_kernel(const float* __restrict__ W1,
                                                      __bf16* __restrict__ out) {
    size_t e = (size_t)blockIdx.x * 256 + threadIdx.x;   // 256*512 elements
    int ki = (int)(e & 31);
    int kt = (int)((e >> 5) & 15);
    int n  = (int)(e >> 9);
    out[((size_t)kt * 256 + n) * 32 + ki] = (__bf16)W1[(size_t)n * H_ + kt * 32 + ki];
}

// Zero c (fp32) and h0 (bf16)
__global__ __launch_bounds__(256) void zero_state_kernel(float* __restrict__ c,
                                                         __bf16* __restrict__ h0) {
    size_t i = (size_t)blockIdx.x * 256 + threadIdx.x;   // B*H elements
    c[i] = 0.0f;
    h0[i] = (__bf16)0.0f;
}

// ---------------------------------------------------------------------------
// One LSTM timestep: gates GEMM (bf16 WMMA, fp32 acc) + fused nonlinear update.
// Grid: (16 m-tiles of 64 rows, 16 n-tiles of 32 gate-cols), 256 threads.
// Wave w: gate = w&3, m-pair p = w>>2; computes a 2x2 grid of 16x16 tiles.
// k-loop split: x half (fp32 load + cvt_pk_bf16 + ds_store), then h half
// (global_load_async_to_lds_b128 + s_wait_asynccnt). A tile double-buffered,
// one barrier per iteration.
// ---------------------------------------------------------------------------
__global__ __launch_bounds__(256)
void lstm_step_kernel(const float* __restrict__ x,      // [B,T,D]
                      const __bf16* __restrict__ wgp,   // packed [32][2048][32]
                      const float* __restrict__ bg,     // [4H]
                      const __bf16* __restrict__ hprev, // [B,H] bf16
                      __bf16* __restrict__ hnext,       // [B,H] bf16
                      float* __restrict__ hout,         // [B,H] fp32 (d_out)
                      float* __restrict__ cbuf,         // [B,H] fp32 (d_out)
                      int t) {
    __shared__ __bf16 Alds[2][64][40];    // double-buffered 64 x 32 K tile
    __shared__ float  G[4][64][33];       // 4 gates x 64 m x 32 n

    const int tid  = threadIdx.x;
    const int lane = tid & 31;
    const int w    = tid >> 5;
    const int g    = w & 3;
    const int p    = w >> 2;              // 0..1
    const int mbase = blockIdx.x * 64;
    const int nbase = blockIdx.y * 32;

    v8f acc00 = {}, acc01 = {}, acc10 = {}, acc11 = {};

    const int lm = tid >> 2;              // coop-stage row 0..63
    const int lk = (tid & 3) * 8;         // coop-stage k 0,8,16,24

    const int ar0 = p * 32 + (lane & 15); // first A-frag row
    const int ak  = (lane >> 4) * 8;      // A-frag k base (0 or 8)
    const int kb  = (lane >> 4) * 8;      // B-frag k base (0 or 8)
    const int nA  = g * H_ + nbase + (lane & 15);

    const float*  xrow = x + ((size_t)(mbase + lm) * T_ + t) * D_ + lk;
    const __bf16* hrow = hprev + (size_t)(mbase + lm) * H_ + lk;

    // ---------------- x half: K = 0..511 (fp32 -> bf16 convert path) -------
    for (int kt = 0; kt < 16; ++kt) {
        __bf16* abuf = &Alds[kt & 1][0][0];
        {
            f32x4 f0 = *(const f32x4*)(xrow + kt * 32);
            f32x4 f1 = *(const f32x4*)(xrow + kt * 32 + 4);
            bf16x4 h0 = { (__bf16)f0.x, (__bf16)f0.y, (__bf16)f0.z, (__bf16)f0.w };
            bf16x4 h1 = { (__bf16)f1.x, (__bf16)f1.y, (__bf16)f1.z, (__bf16)f1.w };
            *(bf16x4*)(abuf + lm * 40 + lk)     = h0;
            *(bf16x4*)(abuf + lm * 40 + lk + 4) = h1;
        }
        __syncthreads();

        Frag a0, a1;
        a0.q[0] = *(const u32x4*)(abuf + (size_t)ar0 * 40 + ak);
        a0.q[1] = *(const u32x4*)(abuf + (size_t)ar0 * 40 + ak + 16);
        a1.q[0] = *(const u32x4*)(abuf + (size_t)(ar0 + 16) * 40 + ak);
        a1.q[1] = *(const u32x4*)(abuf + (size_t)(ar0 + 16) * 40 + ak + 16);

        const __bf16* bp = wgp + ((size_t)kt * G4H + nA) * 32 + kb;
        Frag b0, b1;
        b0.q[0] = *(const u32x4*)bp;           b0.q[1] = *(const u32x4*)(bp + 16);
        b1.q[0] = *(const u32x4*)(bp + 512);   b1.q[1] = *(const u32x4*)(bp + 512 + 16);

        acc00 = __builtin_amdgcn_wmma_f32_16x16x32_bf16(false, a0.v, false, b0.v,
                                                        (short)0, acc00, false, false);
        acc01 = __builtin_amdgcn_wmma_f32_16x16x32_bf16(false, a0.v, false, b1.v,
                                                        (short)0, acc01, false, false);
        acc10 = __builtin_amdgcn_wmma_f32_16x16x32_bf16(false, a1.v, false, b0.v,
                                                        (short)0, acc10, false, false);
        acc11 = __builtin_amdgcn_wmma_f32_16x16x32_bf16(false, a1.v, false, b1.v,
                                                        (short)0, acc11, false, false);
    }

    // ---------------- h half: K = 512..1023 (async global->LDS path) -------
    for (int kt = 16; kt < 32; ++kt) {
        __bf16* abuf = &Alds[kt & 1][0][0];
        async_load_b128(lds_offset(abuf + lm * 40 + lk), hrow + (kt - 16) * 32);
        wait_asynccnt0();
        __syncthreads();

        Frag a0, a1;
        a0.q[0] = *(const u32x4*)(abuf + (size_t)ar0 * 40 + ak);
        a0.q[1] = *(const u32x4*)(abuf + (size_t)ar0 * 40 + ak + 16);
        a1.q[0] = *(const u32x4*)(abuf + (size_t)(ar0 + 16) * 40 + ak);
        a1.q[1] = *(const u32x4*)(abuf + (size_t)(ar0 + 16) * 40 + ak + 16);

        const __bf16* bp = wgp + ((size_t)kt * G4H + nA) * 32 + kb;
        Frag b0, b1;
        b0.q[0] = *(const u32x4*)bp;           b0.q[1] = *(const u32x4*)(bp + 16);
        b1.q[0] = *(const u32x4*)(bp + 512);   b1.q[1] = *(const u32x4*)(bp + 512 + 16);

        acc00 = __builtin_amdgcn_wmma_f32_16x16x32_bf16(false, a0.v, false, b0.v,
                                                        (short)0, acc00, false, false);
        acc01 = __builtin_amdgcn_wmma_f32_16x16x32_bf16(false, a0.v, false, b1.v,
                                                        (short)0, acc01, false, false);
        acc10 = __builtin_amdgcn_wmma_f32_16x16x32_bf16(false, a1.v, false, b0.v,
                                                        (short)0, acc10, false, false);
        acc11 = __builtin_amdgcn_wmma_f32_16x16x32_bf16(false, a1.v, false, b1.v,
                                                        (short)0, acc11, false, false);
    }

    // ---- spill gate accumulators to LDS ----
    {
        const int mr = p * 32 + (lane >> 4) * 8;
        const int nc = lane & 15;
#pragma unroll
        for (int v = 0; v < 8; ++v) {
            G[g][mr + v][nc]           = acc00[v];
            G[g][mr + v][nc + 16]      = acc01[v];
            G[g][mr + 16 + v][nc]      = acc10[v];
            G[g][mr + 16 + v][nc + 16] = acc11[v];
        }
    }
    __syncthreads();

    // ---- fused LSTM nonlinearity + state update (8 elems / thread) ----
    {
        const int m  = tid >> 2;              // 0..63
        const int n0 = (tid & 3) * 8;
        const int brow = mbase + m;
#pragma unroll
        for (int i = 0; i < 8; ++i) {
            const int n = n0 + i;
            const int j = nbase + n;
            float f  = sigmoidf_(G[0][m][n] + bg[j]);
            float in = sigmoidf_(G[1][m][n] + bg[H_ + j]);
            float o  = sigmoidf_(G[2][m][n] + bg[2 * H_ + j]);
            float cd = tanhf    (G[3][m][n] + bg[3 * H_ + j]);
            size_t idx = (size_t)brow * H_ + j;
            float cn = f * cbuf[idx] + in * cd;
            float hn = o * tanhf(cn);
            cbuf[idx]  = cn;
            hout[idx]  = hn;
            hnext[idx] = (__bf16)hn;
        }
    }
}

// ---------------------------------------------------------------------------
// Output head: y = relu(h @ W1^T + b1) @ W2^T + b2, fused.
// Grid: 64 WGs (16 batch rows each), 256 threads; wave w -> cols [w*32,w*32+32)
// ---------------------------------------------------------------------------
__global__ __launch_bounds__(256)
void lstm_head_kernel(const __bf16* __restrict__ hcur,  // [B,H] bf16
                      const __bf16* __restrict__ w1p,   // packed [16][256][32]
                      const float* __restrict__ b1,     // [256]
                      const float* __restrict__ w2,     // [256]
                      const float* __restrict__ b2,     // [1]
                      float* __restrict__ ys,           // [B,T]
                      int t) {
    __shared__ float red[16];
    const int tid  = threadIdx.x;
    const int lane = tid & 31;
    const int w    = tid >> 5;
    const int bb   = blockIdx.x * 16;

    if (tid < 16) red[tid] = 0.0f;
    __syncthreads();

    v8f acc0 = {}; v8f acc1 = {};
    const int ar = bb + (lane & 15);
    const int ak = (lane >> 4) * 8;
    const int kb = (lane >> 4) * 8;
    const int n0 = w * 32 + (lane & 15);

    for (int kt = 0; kt < 16; ++kt) {
        Frag a;
        const __bf16* ap = hcur + (size_t)ar * H_ + kt * 32 + ak;
        a.q[0] = *(const u32x4*)ap;
        a.q[1] = *(const u32x4*)(ap + 16);

        const __bf16* bp = w1p + ((size_t)kt * 256 + n0) * 32 + kb;
        Frag b0; b0.q[0] = *(const u32x4*)bp;       b0.q[1] = *(const u32x4*)(bp + 16);
        acc0 = __builtin_amdgcn_wmma_f32_16x16x32_bf16(false, a.v, false, b0.v,
                                                       (short)0, acc0, false, false);
        const __bf16* bq = bp + 16 * 32;
        Frag b1; b1.q[0] = *(const u32x4*)bq;       b1.q[1] = *(const u32x4*)(bq + 16);
        acc1 = __builtin_amdgcn_wmma_f32_16x16x32_bf16(false, a.v, false, b1.v,
                                                       (short)0, acc1, false, false);
    }

    // bias + relu + W2 dot, then cross-wave reduce via LDS float atomics
    const int na = w * 32 + (lane & 15);
    const int nb = na + 16;
    const float w2a = w2[na], w2b = w2[nb];
    const float b1a = b1[na], b1b = b1[nb];
    const int mrow = (lane >> 4) * 8;
#pragma unroll
    for (int v = 0; v < 8; ++v) {
        float s = fmaxf(acc0[v] + b1a, 0.0f) * w2a + fmaxf(acc1[v] + b1b, 0.0f) * w2b;
        atomicAdd(&red[mrow + v], s);
    }
    __syncthreads();
    if (tid < 16) ys[(size_t)(bb + tid) * T_ + t] = red[tid] + b2[0];
}

// ---------------------------------------------------------------------------
extern "C" void kernel_launch(void* const* d_in, const int* in_sizes, int n_in,
                              void* d_out, int out_size, void* d_ws, size_t ws_size,
                              hipStream_t stream) {
    (void)in_sizes; (void)n_in; (void)out_size; (void)ws_size;
    const float* x  = (const float*)d_in[0];
    const float* Wg = (const float*)d_in[1];
    const float* bg = (const float*)d_in[2];
    const float* W1 = (const float*)d_in[3];
    const float* b1 = (const float*)d_in[4];
    const float* W2 = (const float*)d_in[5];
    const float* b2 = (const float*)d_in[6];

    float* out  = (float*)d_out;
    float* ys   = out;                              // [B,T]
    float* hout = out + (size_t)B_ * T_;            // [B,H]
    float* cbuf = hout + (size_t)B_ * H_;           // [B,H]

    char* ws = (char*)d_ws;
    __bf16* wgp  = (__bf16*)ws;                                   // 4 MB
    __bf16* w1p  = (__bf16*)(ws + 4 * 1024 * 1024);               // 256 KB
    __bf16* hbf0 = (__bf16*)(ws + 4 * 1024 * 1024 + 256 * 1024);  // 1 MB
    __bf16* hbf1 = hbf0 + (size_t)B_ * H_;                        // 1 MB

    pack_wg_kernel<<<(G4H * KTOT) / 256, 256, 0, stream>>>(Wg, wgp);
    pack_w1_kernel<<<(256 * H_) / 256, 256, 0, stream>>>(W1, w1p);
    zero_state_kernel<<<(B_ * H_) / 256, 256, 0, stream>>>(cbuf, hbf0);

    for (int t = 0; t < T_; ++t) {
        __bf16* hp = (t & 1) ? hbf1 : hbf0;
        __bf16* hn = (t & 1) ? hbf0 : hbf1;
        lstm_step_kernel<<<dim3(16, 16), 256, 0, stream>>>(x, wgp, bg, hp, hn, hout, cbuf, t);
        lstm_head_kernel<<<64, 256, 0, stream>>>(hn, w1p, b1, W2, b2, ys, t);
    }
}
